// MultiHeadSelfAttention_75797582840319
// MI455X (gfx1250) — compile-verified, matrix operand order checked by reading, not run on hardware
//
#include <hip/hip_runtime.h>

// ---------------------------------------------------------------------------
// Causal multi-head self-attention, bf16 WMMA pipeline for gfx1250 (MI455X).
//   B=2, T=2048, C=1024, H=16, D=64
// Stages (all on `stream`):
//   1. convert x -> bf16          (xb,   [4096,1024])
//   2. transpose Wqkv -> bf16     (wqT,  [3072,1024])
//   3. transpose Wout -> bf16     (woT,  [1024,1024])
//   4. GEMM qkv = xb @ Wqkv  : 4-wave blocks, 128x64 block tile, B panel staged
//      in LDS via global_load_async_to_lds_b128 (double buffered, ASYNCcnt),
//      routed into Q[B,H,T,D], K[B,H,T,D], Vt[B,H,D,T]
//   5. flash attention per (b,h,16-query tile), online softmax, WMMA score & PV
//   6. GEMM out = Y @ Wout  : same structure as 4, f32 store to d_out
// ---------------------------------------------------------------------------

typedef __bf16 bf16;
typedef __attribute__((ext_vector_type(16))) __bf16 v16bf;
typedef __attribute__((ext_vector_type(8)))  float  v8f;

static constexpr int BB = 2;
static constexpr int TT = 2048;
static constexpr int CC = 1024;
static constexpr int HH = 16;
static constexpr int DD = 64;          // head dim
static constexpr int MROWS = BB * TT;  // 4096

// ---- fragment loaders ------------------------------------------------------
// 16-bit A/B fragment (16x32 MxK, 32x16 KxN mirrored): for lane `l`,
// half = l>>4, the 16 bf16 values are rowptr[half*8 + 0..7] and
// rowptr[16 + half*8 + 0..7]  (two 16-byte loads).
__device__ __forceinline__ v16bf load_frag_row(const bf16* rowptr, int hl) {
    union { uint4 u[2]; v16bf v; } f;
    f.u[0] = *(const uint4*)(rowptr + hl * 8);
    f.u[1] = *(const uint4*)(rowptr + 16 + hl * 8);
    return f.v;
}

// B fragment from an LDS slab laid out as rows of 32 bf16 (row = n column idx)
__device__ __forceinline__ v16bf lds_frag(const bf16* slab, int row, int hl) {
    union { uint4 u[2]; v16bf v; } f;
    f.u[0] = *(const uint4*)(slab + row * 32 + hl * 8);
    f.u[1] = *(const uint4*)(slab + row * 32 + 16 + hl * 8);
    return f.v;
}

__device__ __forceinline__ v8f wmma_bf16(v16bf a, v16bf b, v8f c) {
    return __builtin_amdgcn_wmma_f32_16x16x32_bf16(false, a, false, b,
                                                   (short)0, c, false, false);
}

// CDNA5 async global->LDS copy of 16 bytes (tracked by ASYNCcnt)
__device__ __forceinline__ void async_b128(const bf16* g, bf16* l) {
    unsigned lofs = (unsigned)(unsigned long long)l;   // low 32b = LDS offset
    asm volatile("global_load_async_to_lds_b128 %0, %1, off"
                 :: "v"(lofs), "v"((unsigned long long)g)
                 : "memory");
}

// ---- stage 1/2/3: conversions ---------------------------------------------
__global__ void cvt_f32_to_bf16(const float* __restrict__ src,
                                bf16* __restrict__ dst, int n) {
    int i = blockIdx.x * blockDim.x + threadIdx.x;
    if (i < n) dst[i] = (bf16)src[i];
}

// src [K,N] row-major f32  ->  dst [N,K] row-major bf16
__global__ void transpose_to_bf16(const float* __restrict__ src,
                                  bf16* __restrict__ dst, int K, int N) {
    int i = blockIdx.x * blockDim.x + threadIdx.x;
    if (i < K * N) {
        int k = i / N, n = i % N;
        dst[(size_t)n * K + k] = (bf16)src[i];
    }
}

// ---- shared GEMM core ------------------------------------------------------
// Block = 128 threads (4 waves). Block tile 128(M) x 64(N); wave w handles
// rows [m0+32w, m0+32w+32). B panel rows n0..n0+63 staged in LDS, k-slab of 32,
// double buffered via async-to-LDS. Accumulators returned to caller via lambda
// would bloat; instead the two GEMM kernels duplicate the epilogue.
#define GEMM_CORE(A_, WT_, KDIM_)                                              \
    __shared__ bf16 Bs[2][64 * 32];                                            \
    const int tid  = threadIdx.x;                                              \
    const int lane = tid & 31;                                                 \
    const int wid  = tid >> 5;                                                 \
    const int hl   = lane >> 4;                                                \
    const int nl   = lane & 15;                                                \
    const int mt   = blockIdx.x / NT;                                          \
    const int nt   = blockIdx.x % NT;                                          \
    const int n0   = nt * 64;                                                  \
    const int mw   = mt * 128 + wid * 32;                                      \
    const int fr   = tid >> 1;          /* slab fill: row 0..63 */             \
    const int fp   = (tid & 1) * 16;    /* slab fill: 16-elem part */          \
    const bf16* fsrc = WT_ + (size_t)(n0 + fr) * KDIM_ + fp;                   \
    bf16* fdst0 = &Bs[0][fr * 32 + fp];                                        \
    bf16* fdst1 = &Bs[1][fr * 32 + fp];                                        \
    v8f accA0 = {}, accA1 = {}, accA2 = {}, accA3 = {};                        \
    v8f accB0 = {}, accB1 = {}, accB2 = {}, accB3 = {};                        \
    const bf16* arow0 = A_ + (size_t)(mw + nl) * KDIM_;                        \
    const bf16* arow1 = A_ + (size_t)(mw + 16 + nl) * KDIM_;                   \
    const int KSTEPS = KDIM_ / 32;                                             \
    /* prologue: slab 0 */                                                     \
    async_b128(fsrc, fdst0);                                                   \
    async_b128(fsrc + 8, fdst0 + 8);                                           \
    for (int ks = 0; ks < KSTEPS; ++ks) {                                      \
        const int cur = ks & 1;                                                \
        bf16* curbuf = &Bs[cur][0];                                            \
        if (ks + 1 < KSTEPS) {                                                 \
            const bf16* s = fsrc + (ks + 1) * 32;                              \
            bf16* d = (cur ? fdst0 : fdst1);                                   \
            async_b128(s, d);                                                  \
            async_b128(s + 8, d + 8);                                          \
            asm volatile("s_wait_asynccnt 0x2" ::: "memory");                  \
        } else {                                                               \
            asm volatile("s_wait_asynccnt 0x0" ::: "memory");                  \
        }                                                                      \
        __syncthreads();                                                       \
        const int k0 = ks * 32;                                                \
        v16bf a0 = load_frag_row(arow0 + k0, hl);                              \
        v16bf a1 = load_frag_row(arow1 + k0, hl);                              \
        v16bf b0 = lds_frag(curbuf, 0 * 16 + nl, hl);                          \
        v16bf b1 = lds_frag(curbuf, 1 * 16 + nl, hl);                          \
        v16bf b2 = lds_frag(curbuf, 2 * 16 + nl, hl);                          \
        v16bf b3 = lds_frag(curbuf, 3 * 16 + nl, hl);                          \
        accA0 = wmma_bf16(a0, b0, accA0);                                      \
        accB0 = wmma_bf16(a1, b0, accB0);                                      \
        accA1 = wmma_bf16(a0, b1, accA1);                                      \
        accB1 = wmma_bf16(a1, b1, accB1);                                      \
        accA2 = wmma_bf16(a0, b2, accA2);                                      \
        accB2 = wmma_bf16(a1, b2, accB2);                                      \
        accA3 = wmma_bf16(a0, b3, accA3);                                      \
        accB3 = wmma_bf16(a1, b3, accB3);                                      \
        __syncthreads();                                                       \
    }

// ---- stage 4: QKV projection GEMM -----------------------------------------
__global__ void __launch_bounds__(128)
gemm_qkv(const bf16* __restrict__ A, const bf16* __restrict__ WT,
         bf16* __restrict__ Qb, bf16* __restrict__ Kb, bf16* __restrict__ Vt) {
    const int NT = (3 * CC) / 64;  // 48 n-tiles
    GEMM_CORE(A, WT, CC)

    v8f accs[2][4] = {{accA0, accA1, accA2, accA3},
                      {accB0, accB1, accB2, accB3}};
    for (int g = 0; g < 2; ++g) {
        for (int seg = 0; seg < 4; ++seg) {
            for (int r = 0; r < 8; ++r) {
                int m = mw + g * 16 + r + 8 * hl;   // global row (b*T + t)
                int n = n0 + seg * 16 + nl;         // global qkv column [0,3C)
                int b = m / TT, t = m % TT;
                bf16 v = (bf16)accs[g][seg][r];
                if (n < CC) {
                    int h = n / DD, d = n % DD;
                    Qb[(((size_t)(b * HH + h) * TT) + t) * DD + d] = v;
                } else if (n < 2 * CC) {
                    int nn = n - CC;
                    int h = nn / DD, d = nn % DD;
                    Kb[(((size_t)(b * HH + h) * TT) + t) * DD + d] = v;
                } else {
                    int nn = n - 2 * CC;
                    int h = nn / DD, d = nn % DD;
                    Vt[(((size_t)(b * HH + h) * DD) + d) * TT + t] = v;  // V^T
                }
            }
        }
    }
}

// ---- stage 6: output projection GEMM (f32 store) --------------------------
__global__ void __launch_bounds__(128)
gemm_out(const bf16* __restrict__ A, const bf16* __restrict__ WT,
         float* __restrict__ Out) {
    const int NT = CC / 64;  // 16 n-tiles
    GEMM_CORE(A, WT, CC)

    v8f accs[2][4] = {{accA0, accA1, accA2, accA3},
                      {accB0, accB1, accB2, accB3}};
    for (int g = 0; g < 2; ++g)
        for (int seg = 0; seg < 4; ++seg)
            for (int r = 0; r < 8; ++r) {
                int m = mw + g * 16 + r + 8 * hl;
                int n = n0 + seg * 16 + nl;
                Out[(size_t)m * CC + n] = accs[g][seg][r];
            }
}

// ---- stage 5: flash attention ---------------------------------------------
// One wave (32 threads) per (b, h, 16-query tile). Online softmax.
__global__ void __launch_bounds__(32)
attn_kernel(const bf16* __restrict__ Qb, const bf16* __restrict__ Kb,
            const bf16* __restrict__ Vt, bf16* __restrict__ Yb) {
    __shared__ bf16 Pl[16 * 16];

    int blk = blockIdx.x;
    const int qt = blk % (TT / 16); blk /= (TT / 16);
    const int h  = blk % HH;
    const int b  = blk / HH;
    const int q0 = qt * 16;

    const int lane = threadIdx.x & 31;
    const int hl = lane >> 4;
    const int nl = lane & 15;

    const bf16* Qp = Qb + (size_t)(b * HH + h) * TT * DD;
    const bf16* Kp = Kb + (size_t)(b * HH + h) * TT * DD;
    const bf16* Vp = Vt + (size_t)(b * HH + h) * DD * TT;

    // Q fragments for this tile (A-layout: row m = nl)
    v16bf aq0 = load_frag_row(Qp + (size_t)(q0 + nl) * DD + 0, hl);
    v16bf aq1 = load_frag_row(Qp + (size_t)(q0 + nl) * DD + 32, hl);

    v8f acc[4] = {v8f{}, v8f{}, v8f{}, v8f{}};
    float rmax[8], rsum[8];
    for (int r = 0; r < 8; ++r) { rmax[r] = -1e30f; rsum[r] = 0.0f; }
    const float scale = 0.125f;  // 1/sqrt(64)

    for (int kk0 = 0; kk0 <= q0; kk0 += 16) {
        // scores S = Q K^T  (B-fragment: column n = key kk0+nl, contiguous d)
        v16bf bk0 = load_frag_row(Kp + (size_t)(kk0 + nl) * DD + 0, hl);
        v16bf bk1 = load_frag_row(Kp + (size_t)(kk0 + nl) * DD + 32, hl);
        v8f s = {};
        s = wmma_bf16(aq0, bk0, s);
        s = wmma_bf16(aq1, bk1, s);

        float alpha[8];
        for (int r = 0; r < 8; ++r) {
            const int qrow = q0 + r + 8 * hl;
            const int kcol = kk0 + nl;
            float sv = s[r] * scale;
            if (kcol > qrow) sv = -1e30f;        // causal mask
            float vmax = sv;
            for (int off = 1; off < 16; off <<= 1)
                vmax = fmaxf(vmax, __shfl_xor(vmax, off, 32));
            float mnew = fmaxf(rmax[r], vmax);
            float a = __expf(rmax[r] - mnew);
            float p = __expf(sv - mnew);
            float psum = p;
            for (int off = 1; off < 16; off <<= 1)
                psum += __shfl_xor(psum, off, 32);
            rsum[r] = rsum[r] * a + psum;
            rmax[r] = mnew;
            alpha[r] = a;
            Pl[(r + 8 * hl) * 16 + nl] = (bf16)p;  // P row-major in LDS
        }
        for (int seg = 0; seg < 4; ++seg)
            for (int r = 0; r < 8; ++r)
                acc[seg][r] *= alpha[r];

        // wave-internal LDS visibility before re-layout load
        asm volatile("s_wait_dscnt 0x0" ::: "memory");

        // P as A-fragment (K=0..15 valid, K=16..31 zeroed)
        union { uint4 u[2]; v16bf v; } fp;
        fp.u[0] = *(const uint4*)(&Pl[nl * 16 + hl * 8]);
        fp.u[1] = make_uint4(0u, 0u, 0u, 0u);
        v16bf ap = fp.v;

        // y += P @ V : B-fragment column n = d, contiguous over k in Vt
        for (int seg = 0; seg < 4; ++seg) {
            const bf16* vrow = Vp + (size_t)(seg * 16 + nl) * TT + kk0;
            v16bf bv = load_frag_row(vrow, hl);   // k=16..31 half multiplies zeros
            acc[seg] = wmma_bf16(ap, bv, acc[seg]);
        }
    }

    // normalize and store Y[b*T + q, h*D + d] (bf16)
    for (int seg = 0; seg < 4; ++seg) {
        for (int r = 0; r < 8; ++r) {
            int m = q0 + r + 8 * hl;
            float y = acc[seg][r] / rsum[r];
            Yb[((size_t)(b * TT + m)) * CC + h * DD + seg * 16 + nl] = (bf16)y;
        }
    }
}

// ---------------------------------------------------------------------------
extern "C" void kernel_launch(void* const* d_in, const int* in_sizes, int n_in,
                              void* d_out, int out_size, void* d_ws, size_t ws_size,
                              hipStream_t stream) {
    const float* x    = (const float*)d_in[0];
    // d_in[1] = attn_mask (causal, synthesized in-kernel)
    const float* Wqkv = (const float*)d_in[2];
    const float* Wout = (const float*)d_in[3];
    float* out = (float*)d_out;

    // workspace layout (bf16 elements)
    constexpr size_t N_XB   = (size_t)MROWS * CC;        // 4096*1024
    constexpr size_t N_WQT  = (size_t)3 * CC * CC;       // 3072*1024
    constexpr size_t N_WOT  = (size_t)CC * CC;           // 1024*1024
    constexpr size_t N_HEAD = (size_t)BB * HH * TT * DD; // 4M per tensor
    constexpr size_t PAD    = 128;                       // slack for Vt over-read

    bf16* xb  = (bf16*)d_ws;
    bf16* wqT = xb  + N_XB;
    bf16* woT = wqT + N_WQT;
    bf16* Qb  = woT + N_WOT;
    bf16* Kb  = Qb  + N_HEAD;
    bf16* Vt  = Kb  + N_HEAD;
    bf16* Yb  = Vt  + N_HEAD + PAD;
    (void)ws_size; (void)in_sizes; (void)n_in; (void)out_size;

    // 1-3: precision conversion / weight transposes
    {
        int n = (int)N_XB;
        cvt_f32_to_bf16<<<(n + 255) / 256, 256, 0, stream>>>(x, xb, n);
    }
    {
        int n = (int)N_WQT;
        transpose_to_bf16<<<(n + 255) / 256, 256, 0, stream>>>(Wqkv, wqT, CC, 3 * CC);
    }
    {
        int n = (int)N_WOT;
        transpose_to_bf16<<<(n + 255) / 256, 256, 0, stream>>>(Wout, woT, CC, CC);
    }

    // 4: QKV projection — block tile 128x64: (4096/128)*(3072/64) = 1536 blocks
    gemm_qkv<<<(MROWS / 128) * ((3 * CC) / 64), 128, 0, stream>>>(xb, wqT, Qb, Kb, Vt);

    // 5: attention — B*H*(T/16) = 4096 single-wave blocks
    attn_kernel<<<BB * HH * (TT / 16), 32, 0, stream>>>(Qb, Kb, Vt, Yb);

    // 6: output projection — (4096/128)*(1024/64) = 512 blocks
    gemm_out<<<(MROWS / 128) * (CC / 64), 128, 0, stream>>>(Yb, woT, out);
}
